// AddingModel_16363825398533
// MI455X (gfx1250) — compile-verified, matrix operand order checked by reading, not run on hardware
//
#include <hip/hip_runtime.h>
#include <hip/hip_bf16.h>

// ---- problem constants ----
#define Bb 8
#define Ls 8192
#define Hh 128
#define Nn 32
#define Tt 128      // chunk length
#define Cc 64       // number of chunks = Ls/Tt

typedef __attribute__((ext_vector_type(16))) _Float16 v16h;
typedef __attribute__((ext_vector_type(8)))  _Float16 v8h;
typedef __attribute__((ext_vector_type(8)))  float    v8f;

union Frag { v16h v; v8h h2[2]; };

// ------------------- zero pooled accumulator -------------------
__global__ void k_zero(float* __restrict__ p, int n) {
    int i = blockIdx.x * 256 + threadIdx.x;
    if (i < n) p[i] = 0.f;
}

// ------------------- encoder: u[b][h][l] -------------------
__global__ void k_encoder(const float* __restrict__ x, const float* __restrict__ ew,
                          const float* __restrict__ eb, float* __restrict__ u) {
    int g = blockIdx.x * 256 + threadIdx.x;             // B*H*L = 8388608
    int l = g % Ls;
    int h = (g / Ls) % Hh;
    int b = g / (Ls * Hh);
    float x0 = x[(b * Ls + l) * 2 + 0];
    float x1 = x[(b * Ls + l) * 2 + 1];
    u[(b * Hh + h) * Ls + l] = x0 * ew[h] + x1 * ew[Hh + h] + eb[h];
}

// ------------------- convert out_w to f16 -------------------
__global__ void k_cvtw(const float* __restrict__ w, _Float16* __restrict__ w16, int n) {
    int i = blockIdx.x * 256 + threadIdx.x;
    if (i < n) w16[i] = (_Float16)w[i];
}

// ---------- per-head S4D precompute: P, M(Toeplitz), Q16, decay ----------
__global__ void k_precompute(const float* __restrict__ log_dt, const float* __restrict__ log_A_real,
                             const float* __restrict__ A_imag, const float* __restrict__ C_re,
                             const float* __restrict__ C_im,
                             _Float16* __restrict__ P, _Float16* __restrict__ Mk,
                             _Float16* __restrict__ Q16, float* __restrict__ decay) {
    __shared__ float s_dre[Nn], s_dim[Nn], s_kre[Nn], s_kim[Nn], s_kd[Tt];
    int h = blockIdx.x;
    int t = threadIdx.x;                                // 0..127
    float dt = __expf(log_dt[h]);
    if (t < Nn) {
        float Are = -__expf(log_A_real[h * Nn + t]);
        float Aim = A_imag[h * Nn + t];
        float dre = dt * Are, dimg = dt * Aim;
        float er = __expf(dre);
        float ar = er * __cosf(dimg), ai = er * __sinf(dimg);   // a = exp(dt*A)
        // Kc = C * (a - 1) / A   (complex)
        float nre = ar - 1.f, nim = ai;
        float cre = C_re[h * Nn + t], cim = C_im[h * Nn + t];
        float tre = cre * nre - cim * nim;
        float tim = cre * nim + cim * nre;
        float den = Are * Are + Aim * Aim;
        s_kre[t] = (tre * Are + tim * Aim) / den;
        s_kim[t] = (tim * Are - tre * Aim) / den;
        s_dre[t] = dre; s_dim[t] = dimg;
        float rr = __expf((float)Tt * dre);             // decay = a^T
        decay[(h * Nn + t) * 2 + 0] = rr * __cosf((float)Tt * dimg);
        decay[(h * Nn + t) * 2 + 1] = rr * __sinf((float)Tt * dimg);
    }
    __syncthreads();
    float kd = 0.f;
    for (int n = 0; n < Nn; n++) {
        // a^t
        float r = __expf((float)t * s_dre[n]);
        float pre = r * __cosf((float)t * s_dim[n]);
        float pim = r * __sinf((float)t * s_dim[n]);
        float vre = 2.f * (s_kre[n] * pre - s_kim[n] * pim);
        float vim = 2.f * (s_kre[n] * pim + s_kim[n] * pre);
        P[(h * Tt + t) * 64 + n]      = (_Float16)vre;      // multiplies Re(S)
        P[(h * Tt + t) * 64 + 32 + n] = (_Float16)(-vim);   // multiplies Im(S)
        kd += vre;                                          // K[t] = 2 Re(sum Kc a^t)
        // Q column t: a^(T-t)  (rows 0..31 = Re, 32..63 = Im)
        float e = (float)(Tt - t);
        float r2 = __expf(e * s_dre[n]);
        Q16[(h * 64 + n)      * Tt + t] = (_Float16)(r2 * __cosf(e * s_dim[n]));
        Q16[(h * 64 + 32 + n) * Tt + t] = (_Float16)(r2 * __sinf(e * s_dim[n]));
    }
    s_kd[t] = kd;
    __syncthreads();
    for (int tp = 0; tp < Tt; tp++)
        Mk[(h * Tt + t) * Tt + tp] = (t >= tp) ? (_Float16)s_kd[t - tp] : (_Float16)0.f;
}

// ---------- WMMA chunk-boundary states: G = Q(64x128) @ u_chunk(128x16pad) ----------
__global__ void k_gcompute(const float* __restrict__ u, const _Float16* __restrict__ Q16,
                           float* __restrict__ G) {
    __shared__ __align__(16) _Float16 Us[16 * Tt];      // column-major [col][t]
    int ci = blockIdx.x, h = blockIdx.y;
    int tid = threadIdx.x;                              // 128 threads = 4 waves
    for (int i = tid; i < 16 * Tt; i += 128) {
        int col = i / Tt, row = i % Tt;
        float v = (col < Bb) ? u[(col * Hh + h) * Ls + ci * Tt + row] : 0.f;
        Us[i] = (_Float16)v;
    }
    __syncthreads();
    int wave = tid >> 5, lane = tid & 31;               // wave = M-tile (4 x 16 rows)
    v8f acc = {};
    const _Float16* Qh = Q16 + (long)h * 64 * Tt;
    int mrow = wave * 16 + (lane & 15);
    int kg = (lane >> 4) * 8;
    for (int kt = 0; kt < 4; kt++) {
        int kb = kt * 32;
        Frag fa, fb;
        const _Float16* asrc = Qh + mrow * Tt + kb + kg;
        fa.h2[0] = *(const v8h*)asrc;
        fa.h2[1] = *(const v8h*)(asrc + 16);
        const _Float16* bsrc = &Us[(lane & 15) * Tt + kb + kg];
        fb.h2[0] = *(const v8h*)bsrc;
        fb.h2[1] = *(const v8h*)(bsrc + 16);
        acc = __builtin_amdgcn_wmma_f32_16x16x32_f16(false, fa.v, false, fb.v,
                                                     (short)0, acc, false, false);
    }
    int ncol = lane & 15;
    if (ncol < Bb) {
        for (int v = 0; v < 8; v++) {
            int row = wave * 16 + v + 8 * (lane >> 4);
            G[((h * Cc + ci) * 64 + row) * Bb + ncol] = acc[v];
        }
    }
}

// ---------- sequential (cheap) inter-chunk state scan, in place: G -> entering state S ----------
__global__ void k_scan(float* __restrict__ G, const float* __restrict__ decay) {
    int g = blockIdx.x * 256 + threadIdx.x;             // H*N*B = 32768
    if (g >= Hh * Nn * Bb) return;
    int b = g & 7;
    int n = (g >> 3) & 31;
    int h = g >> 8;
    float dre = decay[(h * Nn + n) * 2 + 0];
    float dimg = decay[(h * Nn + n) * 2 + 1];
    float sre = 0.f, sim = 0.f;
    for (int c = 0; c < Cc; c++) {
        int base = (h * Cc + c) * 64;
        float* pr = &G[(base + n) * Bb + b];
        float* pi = &G[(base + 32 + n) * Bb + b];
        float gre = *pr, gim = *pi;
        *pr = sre; *pi = sim;                           // entering state of chunk c
        float nre = dre * sre - dimg * sim + gre;       // S_{c+1} = a^T S_c + G_c
        float nim = dre * sim + dimg * sre + gim;
        sre = nre; sim = nim;
    }
}

// ---------- WMMA conv: Y = [P | M](128x192) @ [S; u](192x16), fused skip+GELU ----------
__global__ void k_conv(const float* __restrict__ u, const float* __restrict__ G,
                       const _Float16* __restrict__ P, const _Float16* __restrict__ Mk,
                       const float* __restrict__ Dvec, _Float16* __restrict__ y16) {
    __shared__ __align__(16) _Float16 rhs[16 * 192];    // column-major [col][row]
    int ci = blockIdx.x, h = blockIdx.y;
    int tid = threadIdx.x;
    for (int i = tid; i < 16 * 192; i += 256) {
        int col = i / 192, row = i % 192;
        float v = 0.f;
        if (col < Bb) {
            if (row < 64) v = G[((h * Cc + ci) * 64 + row) * Bb + col];
            else          v = u[(col * Hh + h) * Ls + ci * Tt + (row - 64)];
        }
        rhs[i] = (_Float16)v;
    }
    __syncthreads();
    int wave = tid >> 5, lane = tid & 31;               // wave = M-tile (8 waves)
    v8f acc = {};
    const _Float16* Ph = P + (long)h * Tt * 64;
    const _Float16* Mh = Mk + (long)h * Tt * Tt;
    int mrow = wave * 16 + (lane & 15);
    int kg = (lane >> 4) * 8;
    for (int kt = 0; kt < 6; kt++) {
        int kb = kt * 32;
        Frag fa, fb;
        const _Float16* asrc = (kb < 64) ? (Ph + mrow * 64 + kb + kg)
                                         : (Mh + mrow * Tt + (kb - 64) + kg);
        fa.h2[0] = *(const v8h*)asrc;
        fa.h2[1] = *(const v8h*)(asrc + 16);
        const _Float16* bsrc = &rhs[(lane & 15) * 192 + kb + kg];
        fb.h2[0] = *(const v8h*)bsrc;
        fb.h2[1] = *(const v8h*)(bsrc + 16);
        acc = __builtin_amdgcn_wmma_f32_16x16x32_f16(false, fa.v, false, fb.v,
                                                     (short)0, acc, false, false);
    }
    int ncol = lane & 15;
    if (ncol < Bb) {
        float Dh = Dvec[h];
        int b = ncol;
        for (int v = 0; v < 8; v++) {
            int m = v + 8 * (lane >> 4);
            int l = ci * Tt + wave * 16 + m;
            float uv = u[(b * Hh + h) * Ls + l];
            float yv = acc[v] + Dh * uv;
            float y3 = yv * yv * yv;                    // tanh-approx GELU (jax default)
            float gel = 0.5f * yv * (1.f + tanhf(0.7978845608f * (yv + 0.044715f * y3)));
            y16[((long)b * Ls + l) * Hh + h] = (_Float16)gel;  // [pos][h]: B-matrix of next GEMM
        }
    }
}

// ---------- WMMA projection Z = W(256x128) @ Y(128x16-tile), fused bias+GLU+pool ----------
// B tile (16 pos x 128 h, 4KB) staged to LDS with async global->LDS copies (ASYNCcnt path).
__global__ void k_proj(const _Float16* __restrict__ y16, const _Float16* __restrict__ w16,
                       const float* __restrict__ ob, float* __restrict__ pooled) {
    __shared__ __align__(16) _Float16 bsh[16 * Hh];     // row-major [pos][h]
    __shared__ float zs[256 * 16];
    int tile = blockIdx.x;                              // 4096 tiles of 16 positions
    int tid = threadIdx.x, wave = tid >> 5, lane = tid & 31;
    long posbase = (long)tile * 16;

    // async stage: 256 threads x 16B = 4KB tile
    {
        const _Float16* gsrc = y16 + posbase * Hh + tid * 8;
        unsigned ldsa = (unsigned)(size_t)(&bsh[tid * 8]);
        asm volatile("global_load_async_to_lds_b128 %0, %1, off"
                     :: "v"(ldsa), "v"(gsrc) : "memory");
        asm volatile("s_wait_asynccnt 0x0" ::: "memory");
    }
    __syncthreads();

    int kg = (lane >> 4) * 8;
    v8f acc0 = {}, acc1 = {};
    for (int kt = 0; kt < 4; kt++) {
        int kb = kt * 32;
        Frag fb, fa0, fa1;
        const _Float16* bs = &bsh[(lane & 15) * Hh + kb + kg];
        fb.h2[0] = *(const v8h*)bs; fb.h2[1] = *(const v8h*)(bs + 16);
        const _Float16* as0 = w16 + (wave * 32 + (lane & 15)) * Hh + kb + kg;
        fa0.h2[0] = *(const v8h*)as0; fa0.h2[1] = *(const v8h*)(as0 + 16);
        const _Float16* as1 = as0 + 16 * Hh;
        fa1.h2[0] = *(const v8h*)as1; fa1.h2[1] = *(const v8h*)(as1 + 16);
        acc0 = __builtin_amdgcn_wmma_f32_16x16x32_f16(false, fa0.v, false, fb.v,
                                                      (short)0, acc0, false, false);
        acc1 = __builtin_amdgcn_wmma_f32_16x16x32_f16(false, fa1.v, false, fb.v,
                                                      (short)0, acc1, false, false);
    }
    int ncol = lane & 15;
    for (int v = 0; v < 8; v++) {
        int m = v + 8 * (lane >> 4);
        zs[(wave * 32 + m) * 16 + ncol]      = acc0[v];
        zs[(wave * 32 + 16 + m) * 16 + ncol] = acc1[v];
    }
    __syncthreads();
    if (tid < Hh) {
        int hr = tid;
        int b = (int)(posbase / Ls);                    // whole tile lies in one batch
        float ba = ob[hr], bg = ob[Hh + hr];
        float s = 0.f;
        for (int col = 0; col < 16; col++) {
            float a = zs[hr * 16 + col] + ba;
            float gv = zs[(Hh + hr) * 16 + col] + bg;
            s += a * (1.f / (1.f + __expf(-gv)));
        }
        atomicAdd(&pooled[b * Hh + hr], s);
    }
}

// ---------- decode: out[b] = mean-pooled dot dec_w + dec_b ----------
__global__ void k_decode(const float* __restrict__ pooled, const float* __restrict__ dw,
                         const float* __restrict__ db, float* __restrict__ out) {
    __shared__ float sd[128];
    int tid = threadIdx.x;
    for (int b = 0; b < Bb; b++) {
        __syncthreads();
        sd[tid] = pooled[b * Hh + tid] * dw[tid];
        __syncthreads();
        for (int s = 64; s > 0; s >>= 1) {
            if (tid < s) sd[tid] += sd[tid + s];
            __syncthreads();
        }
        if (tid == 0) out[b] = sd[0] * (1.f / (float)Ls) + db[0];
    }
}

extern "C" void kernel_launch(void* const* d_in, const int* in_sizes, int n_in,
                              void* d_out, int out_size, void* d_ws, size_t ws_size,
                              hipStream_t stream) {
    const float* x          = (const float*)d_in[0];
    const float* enc_w      = (const float*)d_in[1];
    const float* enc_b      = (const float*)d_in[2];
    const float* log_dt     = (const float*)d_in[3];
    const float* log_A_real = (const float*)d_in[4];
    const float* A_imag     = (const float*)d_in[5];
    const float* C_re       = (const float*)d_in[6];
    const float* C_im       = (const float*)d_in[7];
    const float* Dvec       = (const float*)d_in[8];
    const float* out_w      = (const float*)d_in[9];
    const float* out_b      = (const float*)d_in[10];
    const float* dec_w      = (const float*)d_in[11];
    const float* dec_b      = (const float*)d_in[12];
    float* out = (float*)d_out;

    // workspace partition (all offsets multiples of 4096)
    char* ws = (char*)d_ws;
    size_t off = 0;
    float*    u_f32 = (float*)(ws + off);    off += (size_t)Bb * Hh * Ls * 4;        // 32 MB
    float*    G     = (float*)(ws + off);    off += (size_t)Hh * Cc * 64 * Bb * 4;   // 16 MB
    _Float16* P     = (_Float16*)(ws + off); off += (size_t)Hh * Tt * 64 * 2;        // 2 MB
    _Float16* Mk    = (_Float16*)(ws + off); off += (size_t)Hh * Tt * Tt * 2;        // 4 MB
    _Float16* Q16   = (_Float16*)(ws + off); off += (size_t)Hh * 64 * Tt * 2;        // 2 MB
    float*    decay = (float*)(ws + off);    off += (size_t)Hh * Nn * 2 * 4;
    _Float16* y16   = (_Float16*)(ws + off); off += (size_t)Bb * Ls * Hh * 2;        // 16 MB
    _Float16* w16   = (_Float16*)(ws + off); off += (size_t)256 * 128 * 2;
    float*    pooled= (float*)(ws + off);    off += (size_t)Bb * Hh * 4;

    k_zero<<<4, 256, 0, stream>>>(pooled, Bb * Hh);
    k_encoder<<<(Bb * Hh * Ls) / 256, 256, 0, stream>>>(x, enc_w, enc_b, u_f32);
    k_cvtw<<<128, 256, 0, stream>>>(out_w, w16, 256 * 128);
    k_precompute<<<Hh, Tt, 0, stream>>>(log_dt, log_A_real, A_imag, C_re, C_im,
                                        P, Mk, Q16, decay);
    k_gcompute<<<dim3(Cc, Hh), 128, 0, stream>>>(u_f32, Q16, G);
    k_scan<<<(Hh * Nn * Bb) / 256, 256, 0, stream>>>(G, decay);
    k_conv<<<dim3(Cc, Hh), 256, 0, stream>>>(u_f32, G, P, Mk, Dvec, y16);
    k_proj<<<(Bb * Ls) / 16, 256, 0, stream>>>(y16, w16, out_b, pooled);
    k_decode<<<1, 128, 0, stream>>>(pooled, dec_w, dec_b, out);
}
